// TransformerEncoderLayer_63522566308196
// MI455X (gfx1250) — matrix operand order repaired and checked by reading, not compiled, not run on hardware
//
#include <hip/hip_runtime.h>
#include <hip/hip_bf16.h>
#include <math.h>

#define DEV __device__ __forceinline__

typedef __attribute__((ext_vector_type(16))) __bf16 v16bf;
typedef __attribute__((ext_vector_type(8)))  __bf16 v8bf;
typedef __attribute__((ext_vector_type(8)))  float  v8f;
typedef __attribute__((ext_vector_type(4)))  unsigned int v4u;
typedef __attribute__((ext_vector_type(8)))  int v8i;
typedef __attribute__((ext_vector_type(4)))  int v4i;

#if defined(__HIP_DEVICE_COMPILE__) && __has_builtin(__builtin_amdgcn_tensor_load_to_lds)
#define USE_TDM 1
#else
#define USE_TDM 0
#endif

// ---------------------------------------------------------------------------
// WMMA helpers (CDNA5 wave32, V_WMMA_F32_16X16X32_BF16)
// ---------------------------------------------------------------------------
DEV v8f wmma_bf16(v16bf a, v16bf b, v8f c) {
  return __builtin_amdgcn_wmma_f32_16x16x32_bf16(
      /*neg_a=*/false, a, /*neg_b=*/false, b,
      /*c_mod=*/(short)0, c, /*reuse_a=*/false, /*reuse_b=*/false);
}

// A fragment 16x32 (MxK), 16-bit: lanes 0-15 hold M=lane with K {0..7,16..23};
// lanes 16-31 hold M=lane-16 with K {8..15,24..31}. base at (M=0,K=0).
DEV v16bf load_a_frag(const __bf16* base, int ldk) {
  const int l = threadIdx.x & 31;
  const __bf16* row = base + (size_t)(l & 15) * ldk + ((l & 16) ? 8 : 0);
  v8bf lo = *(const v8bf*)(row);
  v8bf hi = *(const v8bf*)(row + 16);
  v16bf a;
#pragma unroll
  for (int i = 0; i < 8; ++i) { a[i] = lo[i]; a[i + 8] = hi[i]; }
  return a;
}

// ---------------------------------------------------------------------------
// Tensor Data Mover: 2D tile (rows x cols of bf16) global -> LDS, row-major.
// D# built per CDNA5 ISA. 6-arg builtin variant
// (clang-23 / therock-10.0 headers): (g0, g1, g2, g3, g4, cpol).
// ---------------------------------------------------------------------------
#if USE_TDM
DEV void tdm_load_2d(unsigned lds_addr, const void* gptr,
                     unsigned tile_d0, unsigned tile_d1,
                     unsigned stride_elems, unsigned tensor_d0,
                     unsigned tensor_d1) {
  unsigned long long ga = (unsigned long long)gptr;
  v4u g0;
  g0[0] = 1u;                                   // count=1 valid descriptor
  g0[1] = lds_addr;                             // lds_addr [63:32]
  g0[2] = (unsigned)(ga & 0xffffffffull);       // global_addr[31:0]
  g0[3] = (unsigned)((ga >> 32) & 0x1ffffffull) // global_addr[56:32]
          | (2u << 30);                         // type=2 ("image")
  v8i g1;
  g1[0] = (int)(1u << 16);                      // wg_mask=0, data_size=1 (2B)
  g1[1] = (int)((tensor_d0 & 0xffffu) << 16);   // tensor_dim0[15:0] @bits63:48
  g1[2] = (int)(((tensor_d0 >> 16) & 0xffffu) | ((tensor_d1 & 0xffffu) << 16));
  g1[3] = (int)(((tensor_d1 >> 16) & 0xffffu) | (tile_d0 << 16)); // tile_dim0
  g1[4] = (int)(tile_d1 & 0xffffu);             // tile_dim1, tile_dim2=0
  g1[5] = (int)stride_elems;                    // tensor_dim0_stride[31:0]
  g1[6] = 0;                                    // stride[47:32], dim1_stride lo
  g1[7] = 0;
  v4i z4; z4[0] = 0; z4[1] = 0; z4[2] = 0; z4[3] = 0;
  v8i z8;
#pragma unroll
  for (int i = 0; i < 8; ++i) z8[i] = 0;
  __builtin_amdgcn_tensor_load_to_lds(g0, g1, z4, z4, z8, 0);
}
// Generic pointer to LDS: low 32 bits of the flat address are the LDS offset
DEV unsigned lds_off(const void* p) {
  return (unsigned)(unsigned long long)p;
}
#endif

// ---------------------------------------------------------------------------
// f32 -> bf16 conversion
// ---------------------------------------------------------------------------
__global__ void cvt_f32_bf16_kernel(const float* __restrict__ src,
                                    __bf16* __restrict__ dst, int n) {
  int i = blockIdx.x * 256 + threadIdx.x;
  if (i < n) dst[i] = (__bf16)src[i];
}

// ---------------------------------------------------------------------------
// LayerNorm (torch-style: unbiased var, eps added to std). D must be 1024.
// ---------------------------------------------------------------------------
__global__ __launch_bounds__(256)
void layernorm_kernel(const float* __restrict__ x, const float* __restrict__ alpha,
                      const float* __restrict__ beta, __bf16* __restrict__ out,
                      int D) {
  const int row = blockIdx.x;
  const float* xr = x + (size_t)row * D;
  __shared__ float red[8];
  float xs[4];
  float s = 0.f;
#pragma unroll
  for (int i = 0; i < 4; ++i) { xs[i] = xr[threadIdx.x + i * 256]; s += xs[i]; }
#pragma unroll
  for (int m = 1; m < 32; m <<= 1) s += __shfl_xor(s, m, 32);
  if ((threadIdx.x & 31) == 0) red[threadIdx.x >> 5] = s;
  __syncthreads();
  float tot = 0.f;
#pragma unroll
  for (int i = 0; i < 8; ++i) tot += red[i];
  const float mean = tot / (float)D;
  float ss = 0.f;
#pragma unroll
  for (int i = 0; i < 4; ++i) { float d = xs[i] - mean; ss += d * d; }
#pragma unroll
  for (int m = 1; m < 32; m <<= 1) ss += __shfl_xor(ss, m, 32);
  __syncthreads();
  if ((threadIdx.x & 31) == 0) red[threadIdx.x >> 5] = ss;
  __syncthreads();
  float tss = 0.f;
#pragma unroll
  for (int i = 0; i < 8; ++i) tss += red[i];
  const float var = tss / (float)(D - 1);
  const float inv = 1.f / (sqrtf(var) + 1e-6f);
  __bf16* orow = out + (size_t)row * D;
#pragma unroll
  for (int i = 0; i < 4; ++i) {
    int c = threadIdx.x + i * 256;
    orow[c] = (__bf16)(alpha[c] * (xs[i] - mean) * inv + beta[c]);
  }
}

// ---------------------------------------------------------------------------
// Tiled bf16 WMMA GEMM: C[M,N] = A[M,K]*B[K,N] + bias (+resid)(ReLU?).
// Block tile 128x128, 8 waves (2x4), wave tile 64x32, K-step 64.
// TDM path: double-buffered LDS, DMA of next tiles overlapped with WMMA.
// OMODE: 0 = f32 out [M,N]; 1 = bf16 out [M,N]; 2 = bf16 out transposed per
// head for attention K: out[((b*H+h)*64+d)*S + key].
// ---------------------------------------------------------------------------
template <bool RELU, int OMODE, bool RESID>
__global__ __launch_bounds__(256)
void gemm_bf16_kernel(const __bf16* __restrict__ A, const __bf16* __restrict__ Bm,
                      const float* __restrict__ bias, const float* __restrict__ resid,
                      float* __restrict__ Cf, __bf16* __restrict__ Cb,
                      int M, int N, int K, int S, int H) {
  __shared__ __attribute__((aligned(32))) __bf16 As[2][128 * 64];
  __shared__ __attribute__((aligned(32))) __bf16 Bs[2][64 * 128];
  const int tid = threadIdx.x;
  const int wave = tid >> 5;
  const int wm = wave >> 2;   // 0..1 : 64 rows
  const int wn = wave & 3;    // 0..3 : 32 cols
  const int m0 = blockIdx.y * 128;
  const int n0 = blockIdx.x * 128;

  v8f acc[4][2];
#pragma unroll
  for (int mf = 0; mf < 4; ++mf)
#pragma unroll
    for (int nf = 0; nf < 2; ++nf)
#pragma unroll
      for (int r = 0; r < 8; ++r) acc[mf][nf][r] = 0.f;

#if USE_TDM
  const bool issuer = (tid < 32);        // wave 0 issues TDM descriptors
  if (issuer) {
    tdm_load_2d(lds_off(As[0]), A + (size_t)m0 * K, 64, 128, K, K, M);
    tdm_load_2d(lds_off(Bs[0]), Bm + n0, 128, 64, N, N, K);
    __builtin_amdgcn_s_wait_tensorcnt(0);
  }
  __syncthreads();
  int buf = 0;
  for (int k0 = 0; k0 < K; k0 += 64) {
    const int nk = k0 + 64;
    if (nk < K && issuer) {   // DMA next tiles while we compute this one
      tdm_load_2d(lds_off(As[buf ^ 1]), A + (size_t)m0 * K + nk, 64, 128, K, K, M);
      tdm_load_2d(lds_off(Bs[buf ^ 1]), Bm + (size_t)nk * N + n0, 128, 64, N, N, K);
    }
#pragma unroll
    for (int kc = 0; kc < 2; ++kc) {
      v16bf af[4];
#pragma unroll
      for (int mf = 0; mf < 4; ++mf)
        af[mf] = load_a_frag(&As[buf][(wm * 64 + mf * 16) * 64 + kc * 32], 64);
#pragma unroll
      for (int nf = 0; nf < 2; ++nf) {
        v16bf bfr;
#pragma unroll
        for (int s = 0; s < 16; ++s)
          bfr[s] = Bs[buf][(kc * 32 + ((s & 8) << 1) + ((tid & 16) >> 1) + (s & 7)) * 128 +
                           wn * 32 + nf * 16 + (tid & 15)];
#pragma unroll
        for (int mf = 0; mf < 4; ++mf)
          acc[mf][nf] = wmma_bf16(af[mf], bfr, acc[mf][nf]);
      }
    }
    if (issuer) __builtin_amdgcn_s_wait_tensorcnt(0);
    __syncthreads();
    buf ^= 1;
  }
#else
  for (int k0 = 0; k0 < K; k0 += 64) {
    // manual staging: 2x v16bf per thread per tile
#pragma unroll
    for (int c = 0; c < 2; ++c) {
      int lin = tid * 2 + c;
      int arow = lin >> 2, acol = (lin & 3) * 16;
      *(v16bf*)(&As[0][arow * 64 + acol]) =
          *(const v16bf*)(A + (size_t)(m0 + arow) * K + k0 + acol);
      int brow = lin >> 3, bcol = (lin & 7) * 16;
      *(v16bf*)(&Bs[0][brow * 128 + bcol]) =
          *(const v16bf*)(Bm + (size_t)(k0 + brow) * N + n0 + bcol);
    }
    __syncthreads();
#pragma unroll
    for (int kc = 0; kc < 2; ++kc) {
      v16bf af[4];
#pragma unroll
      for (int mf = 0; mf < 4; ++mf)
        af[mf] = load_a_frag(&As[0][(wm * 64 + mf * 16) * 64 + kc * 32], 64);
#pragma unroll
      for (int nf = 0; nf < 2; ++nf) {
        v16bf bfr;
#pragma unroll
        for (int s = 0; s < 16; ++s)
          bfr[s] = Bs[0][(kc * 32 + ((s & 8) << 1) + ((tid & 16) >> 1) + (s & 7)) * 128 +
                         wn * 32 + nf * 16 + (tid & 15)];
#pragma unroll
        for (int mf = 0; mf < 4; ++mf)
          acc[mf][nf] = wmma_bf16(af[mf], bfr, acc[mf][nf]);
      }
    }
    __syncthreads();
  }
#endif

  // Epilogue. C-layout: lane = N (0..15), VGPR r -> M = r + 8*(lane>=16)
  const int l = tid & 31;
  const int nn = l & 15;
  const int moff = (l & 16) ? 8 : 0;
#pragma unroll
  for (int mf = 0; mf < 4; ++mf) {
#pragma unroll
    for (int nf = 0; nf < 2; ++nf) {
      const int ncol = n0 + wn * 32 + nf * 16 + nn;
      const float bv = bias[ncol];
#pragma unroll
      for (int r = 0; r < 8; ++r) {
        const int mrow = m0 + wm * 64 + mf * 16 + moff + r;
        float v = acc[mf][nf][r] + bv;
        if (RELU) v = v > 0.f ? v : 0.f;
        if (RESID) v += resid[(size_t)mrow * N + ncol];
        if (OMODE == 0) {
          Cf[(size_t)mrow * N + ncol] = v;
        } else if (OMODE == 1) {
          Cb[(size_t)mrow * N + ncol] = (__bf16)v;
        } else {  // transposed-per-head store for attention K
          const int bb = mrow / S, key = mrow % S;
          const int hh = ncol >> 6, dd = ncol & 63;
          Cb[(((size_t)bb * H + hh) * 64 + dd) * S + key] = (__bf16)v;
        }
      }
    }
  }
}

// ---------------------------------------------------------------------------
// Flash attention. One block per (b, h, 64-query tile), 4 independent waves
// of 16 query rows (no block barriers needed). K is pre-transposed to
// KT[b,h,d,key], so logits B-fragments are per-element global loads, coalesced over keys;
// V is consumed in natural [key, d] layout. Online softmax, scale=0.125.
// ---------------------------------------------------------------------------
__global__ __launch_bounds__(128)
void flash_attn_kernel(const __bf16* __restrict__ Q, const __bf16* __restrict__ KT,
                       const __bf16* __restrict__ V, const int* __restrict__ mask,
                       __bf16* __restrict__ ctx, int B, int S, int H) {
  const int b = blockIdx.z, h = blockIdx.y;
  const int wave = threadIdx.x >> 5;
  const int l = threadIdx.x & 31;
  const int D = H * 64;
  const int q0 = blockIdx.x * 64 + wave * 16;

  __shared__ __attribute__((aligned(32))) __bf16 pL[4][16 * 64];  // per-wave P

  const __bf16* ktb = KT + ((size_t)(b * H + h) * 64) * S;  // [d][key]

  v16bf qa[2];
  {
    const __bf16* qbase = Q + ((size_t)(b * S) + q0) * D + h * 64;
#pragma unroll
    for (int c = 0; c < 2; ++c) qa[c] = load_a_frag(qbase + c * 32, D);
  }

  float mrow[8], lrow[8];
  v8f o[4];
#pragma unroll
  for (int r = 0; r < 8; ++r) { mrow[r] = -1e30f; lrow[r] = 0.f; }
#pragma unroll
  for (int j = 0; j < 4; ++j)
#pragma unroll
    for (int r = 0; r < 8; ++r) o[j][r] = 0.f;

  const int moff = (l & 16) ? 8 : 0;
  const int nn = l & 15;

  for (int kb0 = 0; kb0 < S; kb0 += 64) {
    // logits tile S = Q * K^T (16 x 64): 4 n-frags x 2 K-chunks over Dh
    v8f sfr[4];
#pragma unroll
    for (int nf = 0; nf < 4; ++nf) {
      v8f scc;
#pragma unroll
      for (int r = 0; r < 8; ++r) scc[r] = 0.f;
#pragma unroll
      for (int c = 0; c < 2; ++c) {
        v16bf bfr;
#pragma unroll
        for (int s = 0; s < 16; ++s)
          bfr[s] = ktb[(size_t)(c * 32 + ((s & 8) << 1) + ((l & 16) >> 1) + (s & 7)) * S + kb0 + nf * 16 + (l & 15)];
        scc = wmma_bf16(qa[c], bfr, scc);
      }
      sfr[nf] = scc;
    }

    // scale + mask + per-row block max
    float bmax[8];
#pragma unroll
    for (int r = 0; r < 8; ++r) bmax[r] = -1e30f;
#pragma unroll
    for (int nf = 0; nf < 4; ++nf) {
      const int key = kb0 + nf * 16 + nn;
#pragma unroll
      for (int r = 0; r < 8; ++r) {
        const int qrow = q0 + moff + r;
        float v = sfr[nf][r] * 0.125f;
        if (mask[((size_t)b * S + qrow) * S + key] == 0) v = -1e9f;
        sfr[nf][r] = v;
        bmax[r] = fmaxf(bmax[r], v);
      }
    }
#pragma unroll
    for (int r = 0; r < 8; ++r) {
#pragma unroll
      for (int mxo = 1; mxo < 16; mxo <<= 1)
        bmax[r] = fmaxf(bmax[r], __shfl_xor(bmax[r], mxo, 16));
      const float nm = fmaxf(mrow[r], bmax[r]);
      const float corr = __expf(mrow[r] - nm);
      lrow[r] *= corr;
      mrow[r] = nm;
#pragma unroll
      for (int j = 0; j < 4; ++j) o[j][r] *= corr;
    }

    // P = exp(s - m) -> LDS (A-layout reload), accumulate row sums
    float psum[8];
#pragma unroll
    for (int r = 0; r < 8; ++r) psum[r] = 0.f;
#pragma unroll
    for (int nf = 0; nf < 4; ++nf) {
#pragma unroll
      for (int r = 0; r < 8; ++r) {
        float p = __expf(sfr[nf][r] - mrow[r]);
        psum[r] += p;
        pL[wave][(moff + r) * 64 + nf * 16 + nn] = (__bf16)p;
      }
    }
#pragma unroll
    for (int r = 0; r < 8; ++r) {
#pragma unroll
      for (int mxo = 1; mxo < 16; mxo <<= 1)
        psum[r] += __shfl_xor(psum[r], mxo, 16);
      lrow[r] += psum[r];
    }

    // O += P(16x64) * V(64x64); V already K x N in memory
#pragma unroll
    for (int c = 0; c < 2; ++c) {
      v16bf pa = load_a_frag(&pL[wave][0] + c * 32, 64);
#pragma unroll
      for (int j = 0; j < 4; ++j) {
        v16bf vb;
#pragma unroll
        for (int s = 0; s < 16; ++s)
          vb[s] = V[((size_t)(b * S) + kb0 + c * 32 + ((s & 8) << 1) + ((l & 16) >> 1) + (s & 7)) * D +
                    h * 64 + j * 16 + (l & 15)];
        o[j] = wmma_bf16(pa, vb, o[j]);
      }
    }
  }

  // finalize: ctx[b,q,h,d] = O / l
#pragma unroll
  for (int j = 0; j < 4; ++j) {
#pragma unroll
    for (int r = 0; r < 8; ++r) {
      const int qrow = q0 + moff + r;
      float v = o[j][r] / lrow[r];
      ctx[((size_t)(b * S) + qrow) * D + h * 64 + j * 16 + nn] = (__bf16)v;
    }
  }
}

// ---------------------------------------------------------------------------
// Launcher
// ---------------------------------------------------------------------------
extern "C" void kernel_launch(void* const* d_in, const int* in_sizes, int n_in,
                              void* d_out, int out_size, void* d_ws, size_t ws_size,
                              hipStream_t stream) {
  (void)in_sizes; (void)n_in; (void)out_size; (void)ws_size;
  const int B = 2, S = 2048, D = 1024, H = 16, F = 2048;
  const int M = B * S;  // 4096 rows

  const float* inp    = (const float*)d_in[0];
  const int*   mask   = (const int*)  d_in[1];
  const float* wq     = (const float*)d_in[2];
  const float* bq     = (const float*)d_in[3];
  const float* wk     = (const float*)d_in[4];
  const float* bk     = (const float*)d_in[5];
  const float* wv     = (const float*)d_in[6];
  const float* bv     = (const float*)d_in[7];
  const float* wo     = (const float*)d_in[8];
  const float* bo     = (const float*)d_in[9];
  const float* w1     = (const float*)d_in[10];
  const float* b1     = (const float*)d_in[11];
  const float* w2     = (const float*)d_in[12];
  const float* b2     = (const float*)d_in[13];
  const float* alpha1 = (const float*)d_in[14];
  const float* beta1  = (const float*)d_in[15];
  const float* alpha2 = (const float*)d_in[16];
  const float* beta2  = (const float*)d_in[17];
  float* out = (float*)d_out;

  char* ws = (char*)d_ws;
  const size_t MB = 1u << 20;
  __bf16* WQ  = (__bf16*)(ws + 0 * MB);
  __bf16* WK  = (__bf16*)(ws + 2 * MB);
  __bf16* WV  = (__bf16*)(ws + 4 * MB);
  __bf16* WO  = (__bf16*)(ws + 6 * MB);
  __bf16* W1  = (__bf16*)(ws + 8 * MB);
  __bf16* W2  = (__bf16*)(ws + 12 * MB);
  __bf16* XN  = (__bf16*)(ws + 16 * MB);   // ln1 then ln2 output
  __bf16* QB  = (__bf16*)(ws + 24 * MB);
  __bf16* KTB = (__bf16*)(ws + 32 * MB);   // K stored [b,h,d,key]
  __bf16* VB  = (__bf16*)(ws + 40 * MB);
  __bf16* CTX = (__bf16*)(ws + 48 * MB);
  float*  X1  = (float*) (ws + 56 * MB);
  __bf16* HB  = QB;                        // reuse QB+KTB (16MB) after attention

  cvt_f32_bf16_kernel<<<(D * D + 255) / 256, 256, 0, stream>>>(wq, WQ, D * D);
  cvt_f32_bf16_kernel<<<(D * D + 255) / 256, 256, 0, stream>>>(wk, WK, D * D);
  cvt_f32_bf16_kernel<<<(D * D + 255) / 256, 256, 0, stream>>>(wv, WV, D * D);
  cvt_f32_bf16_kernel<<<(D * D + 255) / 256, 256, 0, stream>>>(wo, WO, D * D);
  cvt_f32_bf16_kernel<<<(D * F + 255) / 256, 256, 0, stream>>>(w1, W1, D * F);
  cvt_f32_bf16_kernel<<<(F * D + 255) / 256, 256, 0, stream>>>(w2, W2, F * D);

  layernorm_kernel<<<M, 256, 0, stream>>>(inp, alpha1, beta1, XN, D);

  dim3 gD(D / 128, M / 128);
  gemm_bf16_kernel<false, 1, false><<<gD, 256, 0, stream>>>(
      XN, WQ, bq, nullptr, nullptr, QB, M, D, D, S, H);
  gemm_bf16_kernel<false, 2, false><<<gD, 256, 0, stream>>>(
      XN, WK, bk, nullptr, nullptr, KTB, M, D, D, S, H);   // transposed store
  gemm_bf16_kernel<false, 1, false><<<gD, 256, 0, stream>>>(
      XN, WV, bv, nullptr, nullptr, VB, M, D, D, S, H);

  flash_attn_kernel<<<dim3(S / 64, H, B), 128, 0, stream>>>(
      QB, KTB, VB, mask, CTX, B, S, H);

  gemm_bf16_kernel<false, 0, true><<<gD, 256, 0, stream>>>(
      CTX, WO, bo, inp, X1, nullptr, M, D, D, S, H);

  layernorm_kernel<<<M, 256, 0, stream>>>(X1, alpha2, beta2, XN, D);

  gemm_bf16_kernel<true, 1, false><<<dim3(F / 128, M / 128), 256, 0, stream>>>(
      XN, W1, b1, nullptr, nullptr, HB, M, F, D, S, H);

  gemm_bf16_kernel<false, 0, true><<<gD, 256, 0, stream>>>(
      HB, W2, b2, X1, out, nullptr, M, D, F, S, H);
}
